// BotGAT_GCN_Ensemble_68693706932913
// MI455X (gfx1250) — matrix operand order repaired and checked by reading, not compiled, not run on hardware
//
#include <hip/hip_runtime.h>
#include <hip/hip_bf16.h>
#include <math.h>

// ---------------------------------------------------------------------------
// BotGAT + GCN ensemble for MI455X (gfx1250, wave32, WMMA).
// Dense GEMMs: v_wmma_f32_16x16x32_bf16 (fp32 accumulate), LDS-staged tiles,
// vectorized b128 fragment loads matching the CDNA5 16-bit WMMA layouts.
// Irregular edge softmax / aggregation: L2 float atomics.
// ---------------------------------------------------------------------------

typedef __attribute__((ext_vector_type(16))) __bf16 v16bf;
typedef __attribute__((ext_vector_type(8)))  __bf16 v8bf;
typedef __attribute__((ext_vector_type(4)))  __bf16 v4bf;
typedef __attribute__((ext_vector_type(8)))  float  v8f;

#define HID 256
#define LDA 40   // LDS row stride (elems) for A tile: 80B -> 16B-aligned subchunks
#define LDB 40   // LDS row stride (elems) for Bt tile

// ---------------------------- WMMA GEMM ------------------------------------
// O[M x Kout] = act( A[M x Kin] * W[Kin x Kout] + bias ),  act = leaky_relu(slope)
// (slope = 1.0 -> identity).  Output written at O[m*ldo + col_ofs + n].
// Block: 256 threads = 8 waves; block tile 64(M) x 64(N); K stepped by 32.
// Wave (wr,wc): wr=wave&3 -> rows wr*16, wc=wave>>2 -> cols wc*32 (2 wmma).
__global__ __launch_bounds__(256)
void bg_wmma_gemm(const float* __restrict__ A, int lda,
                  const float* __restrict__ W, int Kin, int Kout,
                  const float* __restrict__ bias,
                  float* __restrict__ O, int ldo, int col_ofs,
                  int Mtot, float slope)
{
    __shared__ alignas(16) __bf16 sA[64 * LDA];   // [row][k]  64 x 32
    __shared__ alignas(16) __bf16 sBt[64 * LDB];  // [col][k]  64 x 32 (transposed W)

    const int tid  = threadIdx.x;
    const int wave = tid >> 5;
    const int lane = tid & 31;
    const int wr   = wave & 3;           // rows wr*16
    const int wc   = wave >> 2;          // cols wc*32
    const int mBase = blockIdx.x * 64;
    const int nBase = blockIdx.y * 64;

    const int g  = lane >> 4;            // lane half
    const int ml = lane & 15;            // M within 16 (A) / N within 16 (B,C)

    v8f acc0 = {0.f, 0.f, 0.f, 0.f, 0.f, 0.f, 0.f, 0.f};
    v8f acc1 = {0.f, 0.f, 0.f, 0.f, 0.f, 0.f, 0.f, 0.f};

    const bool vecOK = ((lda & 3) == 0) && ((Kout & 3) == 0);

    for (int k0 = 0; k0 < Kin; k0 += 32) {
        const bool interior = vecOK && (mBase + 64 <= Mtot) &&
                              (k0 + 32 <= Kin) && (nBase + 64 <= Kout);
        if (interior) {
            // ---- A tile: 64x32 floats, float4 loads, paired bf16 converts ----
#pragma unroll
            for (int j = 0; j < 2; ++j) {
                int q  = tid + j * 256;          // 0..511 float4s
                int r  = q >> 3;                 // row 0..63
                int kq = (q & 7) * 4;            // 0..28
                float4 f = *reinterpret_cast<const float4*>(
                    A + (size_t)(mBase + r) * lda + k0 + kq);
                v4bf pk;
                pk[0] = (__bf16)f.x; pk[1] = (__bf16)f.y;
                pk[2] = (__bf16)f.z; pk[3] = (__bf16)f.w;
                *reinterpret_cast<v4bf*>(&sA[r * LDA + kq]) = pk;
            }
            // ---- W tile: 32x64 floats, float4 loads, transposed stores ----
#pragma unroll
            for (int j = 0; j < 2; ++j) {
                int q   = tid + j * 256;         // 0..511 float4s
                int kr  = q >> 4;                // k row 0..31
                int cc4 = (q & 15) * 4;          // col 0..60
                float4 f = *reinterpret_cast<const float4*>(
                    W + (size_t)(k0 + kr) * Kout + nBase + cc4);
                sBt[(cc4 + 0) * LDB + kr] = (__bf16)f.x;
                sBt[(cc4 + 1) * LDB + kr] = (__bf16)f.y;
                sBt[(cc4 + 2) * LDB + kr] = (__bf16)f.z;
                sBt[(cc4 + 3) * LDB + kr] = (__bf16)f.w;
            }
        } else {
            // ---- guarded path: clamp index, load unconditionally, select 0 ----
#pragma unroll
            for (int j = 0; j < 8; ++j) {
                int idx = tid + j * 256;         // A: 64x32 elems
                int r = idx >> 5, k = idx & 31;
                int grow = mBase + r, gk = k0 + k;
                bool ok = (grow < Mtot) && (gk < Kin);
                int rs = ok ? grow : 0, ks = ok ? gk : 0;
                float v = A[(size_t)rs * lda + ks];
                sA[r * LDA + k] = ok ? (__bf16)v : (__bf16)0.f;
            }
#pragma unroll
            for (int j = 0; j < 8; ++j) {
                int idx = tid + j * 256;         // W: 32x64 elems
                int kr = idx >> 6, cc = idx & 63;
                int gk = k0 + kr, gc = nBase + cc;
                bool ok = (gk < Kin) && (gc < Kout);
                int ks = ok ? gk : 0, cs = ok ? gc : 0;
                float v = W[(size_t)ks * Kout + cs];
                sBt[cc * LDB + kr] = ok ? (__bf16)v : (__bf16)0.f;
            }
        }
        __syncthreads();

        // ---- fragments: contiguous b128 chunks per CDNA5 16-bit layouts ----
        // A lane(g,ml): elems 0..7 -> K=g*8..g*8+7 ; elems 8..15 -> K=16+g*8..+7
        const int arow = wr * 16 + ml;
        v8bf alo = *reinterpret_cast<const v8bf*>(&sA[arow * LDA + g * 8]);
        v8bf ahi = *reinterpret_cast<const v8bf*>(&sA[arow * LDA + 16 + g * 8]);
        v16bf af = __builtin_shufflevector(alo, ahi,
            0, 1, 2, 3, 4, 5, 6, 7, 8, 9, 10, 11, 12, 13, 14, 15);

        // B lane(g,ml): elems 0..7 -> K=g*8..g*8+7 ; elems 8..15 -> K=16+g*8..+7, fixed col
        const int bcol0 = (wc * 32 + ml) * LDB;
        v8bf b0lo = *reinterpret_cast<const v8bf*>(&sBt[bcol0 + g * 8]);
        v8bf b0hi = *reinterpret_cast<const v8bf*>(&sBt[bcol0 + 16 + g * 8]);
        v16bf bf0 = __builtin_shufflevector(b0lo, b0hi,
            0, 1, 2, 3, 4, 5, 6, 7, 8, 9, 10, 11, 12, 13, 14, 15);
        const int bcol1 = (wc * 32 + 16 + ml) * LDB;
        v8bf b1lo = *reinterpret_cast<const v8bf*>(&sBt[bcol1 + g * 8]);
        v8bf b1hi = *reinterpret_cast<const v8bf*>(&sBt[bcol1 + 16 + g * 8]);
        v16bf bf1 = __builtin_shufflevector(b1lo, b1hi,
            0, 1, 2, 3, 4, 5, 6, 7, 8, 9, 10, 11, 12, 13, 14, 15);

        acc0 = __builtin_amdgcn_wmma_f32_16x16x32_bf16(
                   false, af, false, bf0, (short)0, acc0, false, false);
        acc1 = __builtin_amdgcn_wmma_f32_16x16x32_bf16(
                   false, af, false, bf1, (short)0, acc1, false, false);
        __syncthreads();
    }

    // ---- store: VGPR r -> M = r (lanes 0-15) / r+8 (lanes 16-31), N = lane%16
    const int mo = g << 3;
#pragma unroll
    for (int t = 0; t < 2; ++t) {
        const v8f acc = t ? acc1 : acc0;
        int Ng = nBase + wc * 32 + t * 16 + ml;
        if (Ng >= Kout) continue;
        float bv = bias ? bias[Ng] : 0.f;
#pragma unroll
        for (int r = 0; r < 8; ++r) {
            int Mg = mBase + wr * 16 + mo + r;
            if (Mg < Mtot) {
                float v = acc[r] + bv;
                v = (v > 0.f) ? v : v * slope;
                O[(size_t)Mg * ldo + col_ofs + Ng] = v;
            }
        }
    }
}

// ---------------------------- helpers --------------------------------------
__global__ void bg_fill_f32(float* p, float v, int n) {
    int i = blockIdx.x * blockDim.x + threadIdx.x;
    if (i < n) p[i] = v;
}
__global__ void bg_fill_u32(unsigned int* p, unsigned int v, int n) {
    int i = blockIdx.x * blockDim.x + threadIdx.x;
    if (i < n) p[i] = v;
}
// out[n][c] = bias[c]   (row-broadcast init for post-aggregation "+ b")
__global__ void bg_row_bias_init(float* out, const float* __restrict__ b, int n, int c) {
    int i = blockIdx.x * blockDim.x + threadIdx.x;
    if (i < n * c) out[i] = b[i % c];
}

__device__ __forceinline__ void bg_edge_sd(const int* __restrict__ ei, int E, int e,
                                           int& s, int& d) {
    if (e < E) { s = ei[e]; d = ei[E + e]; }
    else       { s = d = e - E; }           // appended self loops
}

__device__ __forceinline__ float bg_atomic_max_f32(float* addr, float val) {
    if (val >= 0.f)
        return __int_as_float(atomicMax((int*)addr, __float_as_int(val)));
    else
        return __uint_as_float(atomicMin((unsigned int*)addr, __float_as_uint(val)));
}

// ---------------------------- GAT edge kernels ------------------------------
// a_s[n,h] = sum_c h[n, h*C+c] * att_src[h,c]   (same for a_d)
__global__ void bg_att_scores(const float* __restrict__ h,
                              const float* __restrict__ att_src,
                              const float* __restrict__ att_dst,
                              float* __restrict__ a_s, float* __restrict__ a_d,
                              int n, int H, int C)
{
    int i = blockIdx.x * blockDim.x + threadIdx.x;
    if (i >= n * H) return;
    int node = i / H, hd = i % H;
    const float* hp = h + (size_t)node * HID + hd * C;
    const float* ws = att_src + hd * C;
    const float* wd = att_dst + hd * C;
    float s = 0.f, d = 0.f;
    for (int c = 0; c < C; c += 4) {
        float4 hv = *reinterpret_cast<const float4*>(hp + c);
        float4 sv = *reinterpret_cast<const float4*>(ws + c);
        float4 dv = *reinterpret_cast<const float4*>(wd + c);
        s += hv.x * sv.x + hv.y * sv.y + hv.z * sv.z + hv.w * sv.w;
        d += hv.x * dv.x + hv.y * dv.y + hv.z * dv.z + hv.w * dv.w;
    }
    a_s[i] = s; a_d[i] = d;
}

__global__ void bg_edge_max(const int* __restrict__ ei, int E, int Et,
                            const float* __restrict__ a_s, const float* __restrict__ a_d,
                            float* __restrict__ m, int H)
{
    int i = blockIdx.x * blockDim.x + threadIdx.x;
    if (i >= Et * H) return;
    int e = i / H, hd = i % H;
    int s, d; bg_edge_sd(ei, E, e, s, d);
    float ev = a_s[s * H + hd] + a_d[d * H + hd];
    ev = (ev > 0.f) ? ev : 0.2f * ev;
    bg_atomic_max_f32(&m[d * H + hd], ev);
}

__global__ void bg_edge_expsum(const int* __restrict__ ei, int E, int Et,
                               const float* __restrict__ a_s, const float* __restrict__ a_d,
                               const float* __restrict__ m, float* __restrict__ den, int H)
{
    int i = blockIdx.x * blockDim.x + threadIdx.x;
    if (i >= Et * H) return;
    int e = i / H, hd = i % H;
    int s, d; bg_edge_sd(ei, E, e, s, d);
    float ev = a_s[s * H + hd] + a_d[d * H + hd];
    ev = (ev > 0.f) ? ev : 0.2f * ev;
    atomicAdd(&den[d * H + hd], expf(ev - m[d * H + hd]));
}

// out[dst, :] += alpha(edge,head) * h[src, :]; 64 threads/edge, 4 channels each
__global__ void bg_gat_aggregate(const int* __restrict__ ei, int E, int Et,
                                 const float* __restrict__ h,
                                 const float* __restrict__ a_s, const float* __restrict__ a_d,
                                 const float* __restrict__ m, const float* __restrict__ den,
                                 float* __restrict__ out, int H, int C)
{
    long long i = (long long)blockIdx.x * blockDim.x + threadIdx.x;
    if (i >= (long long)Et * 64) return;
    int e  = (int)(i >> 6);
    int c0 = ((int)i & 63) * 4;
    int hd = c0 / C;
    int s, d; bg_edge_sd(ei, E, e, s, d);
    float ev = a_s[s * H + hd] + a_d[d * H + hd];
    ev = (ev > 0.f) ? ev : 0.2f * ev;
    float alpha = expf(ev - m[d * H + hd]) / (den[d * H + hd] + 1e-16f);
    float4 hv = *reinterpret_cast<const float4*>(h + (size_t)s * HID + c0);
    float* op = out + (size_t)d * HID + c0;
    atomicAdd(op + 0, alpha * hv.x);
    atomicAdd(op + 1, alpha * hv.y);
    atomicAdd(op + 2, alpha * hv.z);
    atomicAdd(op + 3, alpha * hv.w);
}

// ---------------------------- GCN edge kernels ------------------------------
__global__ void bg_degree(const int* __restrict__ ei, int E, int Et, float* __restrict__ deg) {
    int e = blockIdx.x * blockDim.x + threadIdx.x;
    if (e >= Et) return;
    int s, d; bg_edge_sd(ei, E, e, s, d);
    atomicAdd(&deg[d], 1.0f);
}
__global__ void bg_dinv(const float* __restrict__ deg, float* __restrict__ dinv, int n) {
    int i = blockIdx.x * blockDim.x + threadIdx.x;
    if (i < n) dinv[i] = (deg[i] > 0.f) ? rsqrtf(deg[i]) : 0.f;
}
__global__ void bg_gcn_aggregate(const int* __restrict__ ei, int E, int Et,
                                 const float* __restrict__ h,
                                 const float* __restrict__ dinv,
                                 float* __restrict__ out)
{
    long long i = (long long)blockIdx.x * blockDim.x + threadIdx.x;
    if (i >= (long long)Et * 64) return;
    int e  = (int)(i >> 6);
    int c0 = ((int)i & 63) * 4;
    int s, d; bg_edge_sd(ei, E, e, s, d);
    float norm = dinv[s] * dinv[d];
    float4 hv = *reinterpret_cast<const float4*>(h + (size_t)s * HID + c0);
    float* op = out + (size_t)d * HID + c0;
    atomicAdd(op + 0, norm * hv.x);
    atomicAdd(op + 1, norm * hv.y);
    atomicAdd(op + 2, norm * hv.z);
    atomicAdd(op + 3, norm * hv.w);
}

// ---------------------------- host-side orchestration -----------------------
// Input flattening follows jax pytree (sorted-key) order.
enum {
    IN_DES = 0, IN_TWEET, IN_NUM, IN_CAT, IN_EDGE,
    P_CAT_W, P_CAT_B, P_CATG_W, P_CATG_B,
    P_DES_W, P_DES_B, P_DESG_W, P_DESG_B,
    P_ENS_W, P_ENS_B,
    P_GAT1_W, P_GAT1_ADST, P_GAT1_ASRC, P_GAT1_B,
    P_GAT2_W, P_GAT2_ADST, P_GAT2_ASRC, P_GAT2_B,
    P_GCN1_W, P_GCN1_B, P_GCN2_W, P_GCN2_B,
    P_INP_W, P_INP_B, P_INPG_W, P_INPG_B,
    P_NUM_W, P_NUM_B, P_NUMG_W, P_NUMG_B,
    P_OUT1_W, P_OUT1_B, P_OUT1G_W, P_OUT1G_B,
    P_TWEET_W, P_TWEET_B, P_TWEETG_W, P_TWEETG_B
};

static inline int bg_cdiv(long long a, long long b) { return (int)((a + b - 1) / b); }

static void launch_gemm(const float* A, int lda, const float* W, int Kin, int Kout,
                        const float* bias, float* O, int ldo, int col_ofs,
                        int M, float slope, hipStream_t s)
{
    dim3 grid(bg_cdiv(M, 64), bg_cdiv(Kout, 64));
    bg_wmma_gemm<<<grid, 256, 0, s>>>(A, lda, W, Kin, Kout, bias, O, ldo, col_ofs, M, slope);
}

extern "C" void kernel_launch(void* const* d_in, const int* in_sizes, int n_in,
                              void* d_out, int out_size, void* d_ws, size_t ws_size,
                              hipStream_t stream)
{
    const int N  = in_sizes[IN_DES] / 768;
    const int E  = in_sizes[IN_EDGE] / 2;
    const int Et = E + N;

    const float* des   = (const float*)d_in[IN_DES];
    const float* tweet = (const float*)d_in[IN_TWEET];
    const float* numP  = (const float*)d_in[IN_NUM];
    const float* catP  = (const float*)d_in[IN_CAT];
    const int*   edge  = (const int*)d_in[IN_EDGE];
    auto P = [&](int i) { return (const float*)d_in[i]; };

    float* out = (float*)d_out;

    // workspace carve
    float* buf0 = (float*)d_ws;                     // [N,256]
    float* buf1 = buf0 + (size_t)N * HID;
    float* buf2 = buf1 + (size_t)N * HID;
    float* buf3 = buf2 + (size_t)N * HID;
    float* a_s  = buf3 + (size_t)N * HID;           // [N,4]
    float* a_d  = a_s + (size_t)N * 4;
    float* mbuf = a_d + (size_t)N * 4;
    float* den  = mbuf + (size_t)N * 4;
    float* deg  = den + (size_t)N * 4;              // [N]
    float* dinv = deg + N;

    const int T = 256;
    auto gridN = [&](long long n) { return bg_cdiv(n, T); };

    // ---- shared: degree normalization (same graph for both GCN convs) ----
    bg_fill_f32<<<gridN(N), T, 0, stream>>>(deg, 0.f, N);
    bg_degree<<<gridN(Et), T, 0, stream>>>(edge, E, Et, deg);
    bg_dinv<<<gridN(N), T, 0, stream>>>(deg, dinv, N);

    // =========================== GAT branch ===========================
    launch_gemm(des,   768, P(P_DES_W),   768, 64, P(P_DES_B),   buf0, HID, 0,   N, 0.01f, stream);
    launch_gemm(tweet, 768, P(P_TWEET_W), 768, 64, P(P_TWEET_B), buf0, HID, 64,  N, 0.01f, stream);
    launch_gemm(numP,  5,   P(P_NUM_W),   5,   64, P(P_NUM_B),   buf0, HID, 128, N, 0.01f, stream);
    launch_gemm(catP,  3,   P(P_CAT_W),   3,   64, P(P_CAT_B),   buf0, HID, 192, N, 0.01f, stream);
    launch_gemm(buf0, HID, P(P_INP_W), HID, HID, P(P_INP_B), buf1, HID, 0, N, 0.01f, stream);

    // --- gat1: heads=4, C=64 ---
    launch_gemm(buf1, HID, P(P_GAT1_W), HID, HID, nullptr, buf0, HID, 0, N, 1.0f, stream); // h1
    bg_att_scores<<<gridN((long long)N * 4), T, 0, stream>>>(
        buf0, P(P_GAT1_ASRC), P(P_GAT1_ADST), a_s, a_d, N, 4, 64);
    bg_fill_u32<<<gridN(N * 4), T, 0, stream>>>((unsigned int*)mbuf, 0xFF800000u, N * 4);
    bg_fill_f32<<<gridN(N * 4), T, 0, stream>>>(den, 0.f, N * 4);
    bg_row_bias_init<<<gridN((long long)N * HID), T, 0, stream>>>(buf2, P(P_GAT1_B), N, HID);
    bg_edge_max<<<gridN((long long)Et * 4), T, 0, stream>>>(edge, E, Et, a_s, a_d, mbuf, 4);
    bg_edge_expsum<<<gridN((long long)Et * 4), T, 0, stream>>>(edge, E, Et, a_s, a_d, mbuf, den, 4);
    bg_gat_aggregate<<<gridN((long long)Et * 64), T, 0, stream>>>(
        edge, E, Et, buf0, a_s, a_d, mbuf, den, buf2, 4, 64);

    // --- gat2: heads=1, C=256 ---
    launch_gemm(buf2, HID, P(P_GAT2_W), HID, HID, nullptr, buf0, HID, 0, N, 1.0f, stream); // h2
    bg_att_scores<<<gridN(N), T, 0, stream>>>(
        buf0, P(P_GAT2_ASRC), P(P_GAT2_ADST), a_s, a_d, N, 1, 256);
    bg_fill_u32<<<gridN(N), T, 0, stream>>>((unsigned int*)mbuf, 0xFF800000u, N);
    bg_fill_f32<<<gridN(N), T, 0, stream>>>(den, 0.f, N);
    bg_row_bias_init<<<gridN((long long)N * HID), T, 0, stream>>>(buf1, P(P_GAT2_B), N, HID);
    bg_edge_max<<<gridN(Et), T, 0, stream>>>(edge, E, Et, a_s, a_d, mbuf, 1);
    bg_edge_expsum<<<gridN(Et), T, 0, stream>>>(edge, E, Et, a_s, a_d, mbuf, den, 1);
    bg_gat_aggregate<<<gridN((long long)Et * 64), T, 0, stream>>>(
        edge, E, Et, buf0, a_s, a_d, mbuf, den, buf1, 1, 256);

    launch_gemm(buf1, HID, P(P_OUT1_W), HID, HID, P(P_OUT1_B), buf2, HID, 0, N, 0.01f, stream);
    // GAT branch final result: buf2

    // =========================== GCN branch ===========================
    launch_gemm(des,   768, P(P_DESG_W),   768, 64, P(P_DESG_B),   buf0, HID, 0,   N, 0.01f, stream);
    launch_gemm(tweet, 768, P(P_TWEETG_W), 768, 64, P(P_TWEETG_B), buf0, HID, 64,  N, 0.01f, stream);
    launch_gemm(numP,  5,   P(P_NUMG_W),   5,   64, P(P_NUMG_B),   buf0, HID, 128, N, 0.01f, stream);
    launch_gemm(catP,  3,   P(P_CATG_W),   3,   64, P(P_CATG_B),   buf0, HID, 192, N, 0.01f, stream);
    launch_gemm(buf0, HID, P(P_INPG_W), HID, HID, P(P_INPG_B), buf1, HID, 0, N, 0.01f, stream);

    // gcn1
    launch_gemm(buf1, HID, P(P_GCN1_W), HID, HID, nullptr, buf0, HID, 0, N, 1.0f, stream);
    bg_row_bias_init<<<gridN((long long)N * HID), T, 0, stream>>>(buf1, P(P_GCN1_B), N, HID);
    bg_gcn_aggregate<<<gridN((long long)Et * 64), T, 0, stream>>>(edge, E, Et, buf0, dinv, buf1);
    // gcn2
    launch_gemm(buf1, HID, P(P_GCN2_W), HID, HID, nullptr, buf0, HID, 0, N, 1.0f, stream);
    bg_row_bias_init<<<gridN((long long)N * HID), T, 0, stream>>>(buf3, P(P_GCN2_B), N, HID);
    bg_gcn_aggregate<<<gridN((long long)Et * 64), T, 0, stream>>>(edge, E, Et, buf0, dinv, buf3);

    launch_gemm(buf3, HID, P(P_OUT1G_W), HID, HID, P(P_OUT1G_B), buf0, HID, 0, N, 0.01f, stream);
    // GCN branch final result: buf0

    // =========================== ensemble head ===========================
    launch_gemm(buf2, HID, P(P_ENS_W), HID, 2, P(P_ENS_B), out,                  2, 0, N, 1.0f, stream);
    launch_gemm(buf0, HID, P(P_ENS_W), HID, 2, P(P_ENS_B), out + (size_t)N * 2,  2, 0, N, 1.0f, stream);
}
